// CombinedModel_65120294142533
// MI455X (gfx1250) — matrix operand order repaired and checked by reading, not compiled, not run on hardware
//
#include <hip/hip_runtime.h>
#include <hip/hip_fp16.h>
#include <math.h>

typedef __attribute__((ext_vector_type(16))) _Float16 v16h;
typedef __attribute__((ext_vector_type(8)))  _Float16 v8h;
typedef __attribute__((ext_vector_type(8)))  float    v8f;

// Problem constants
#define BB 64
#define SS 512
#define DD 768
#define FF 256
#define LL 21
#define NPAD 32
#define NEGINF (-1.0e30f)

// ---------------------------------------------------------------------------
// L2-normalize rows of text [512,768] and visual [64,768], emit f16.
// One 256-thread block per row.
// ---------------------------------------------------------------------------
__global__ void l2norm_kernel(const float* __restrict__ text,
                              const float* __restrict__ vis,
                              _Float16* __restrict__ t16,
                              _Float16* __restrict__ v16) {
  int row = blockIdx.x;
  int tid = threadIdx.x;
  const float* src;
  _Float16* dst;
  if (row < SS) { src = text + (size_t)row * DD; dst = t16 + (size_t)row * DD; }
  else          { int r = row - SS; src = vis + (size_t)r * DD; dst = v16 + (size_t)r * DD; }

  __shared__ float red[256];
  float ss = 0.f;
  for (int j = tid; j < DD; j += 256) { float x = src[j]; ss += x * x; }
  red[tid] = ss; __syncthreads();
  for (int st = 128; st > 0; st >>= 1) {
    if (tid < st) red[tid] += red[tid + st];
    __syncthreads();
  }
  float nrm = sqrtf(red[0]);
  float sc = 1.0f / fmaxf(nrm, 1e-12f);
  for (int j = tid; j < DD; j += 256) dst[j] = (_Float16)(src[j] * sc);
}

// ---------------------------------------------------------------------------
// Transpose + f16-convert W_fuse [1536,256] into Wt^T [256,768], Wv^T [256,768]
// ---------------------------------------------------------------------------
__global__ void prep_wfuse_kernel(const float* __restrict__ W,
                                  _Float16* __restrict__ WtT,
                                  _Float16* __restrict__ WvT) {
  int k = blockIdx.x;      // 0..1535
  int n = threadIdx.x;     // 0..255
  _Float16 h = (_Float16)W[(size_t)k * FF + n];
  if (k < DD) WtT[(size_t)n * DD + k] = h;
  else        WvT[(size_t)n * DD + (k - DD)] = h;
}

// W_out [256,21] -> WoT [32,256] (zero-padded columns 21..31), f16
__global__ void prep_wout_kernel(const float* __restrict__ W,
                                 _Float16* __restrict__ WoT) {
  int k = blockIdx.x;      // 0..255
  int n = threadIdx.x;     // 0..31
  float v = (n < LL) ? W[(size_t)k * LL + n] : 0.f;
  WoT[(size_t)n * FF + k] = (_Float16)v;
}

// ---------------------------------------------------------------------------
// WMMA GEMM: C[M,Nout] = A[M,K] @ Bt[N,K]^T (+ bias), one wave per 16x16 tile.
// A row-major f16, Bt row-major f16 (B transposed), C f32 with stride Nout.
// K % 32 == 0, M % 16 == 0, N % 16 == 0.
// ISA layouts (CDNA5 ISA):
//   A 16x32 f16 : lane l<16 holds row l, K {0..7,16..23}; lane l+16: K {8..15,24..31}
//   B 32x16 f16 : lane l<16 holds col l, K {0..7,16..23}; lane l+16 holds col l-16, K {8..15,24..31}
//   C 16x16 f32 : lane l holds col (l&15), rows {0..7} (l<16) / {8..15} (l>=16)
// ---------------------------------------------------------------------------
__global__ void wmma_gemm_kernel(const _Float16* __restrict__ A,
                                 const _Float16* __restrict__ Bt,
                                 float* __restrict__ C,
                                 int M, int N, int K, int Nout,
                                 const float* __restrict__ bias) {
  int lane = threadIdx.x;                     // 0..31 (wave32)
  int m0 = blockIdx.x * 16;
  int n0 = blockIdx.y * 16;
  int half = lane >> 4;                       // 0 or 1

  int arow = m0 + (lane & 15);
  int akb  = half * 8;                        // A K sub-base within 32-block
  int bcol = n0 + (lane & 15);
  int bkb  = half * 8;                        // B K sub-base within 32-block

  const _Float16* aRow = A + (size_t)arow * K;
  const _Float16* bRow = Bt + (size_t)bcol * K;

  v8f c = {};
  for (int k0 = 0; k0 < K; k0 += 32) {
    const _Float16* ap = aRow + k0 + akb;
    v8h a0 = *(const v8h*)(ap);               // K {kb..kb+7}
    v8h a1 = *(const v8h*)(ap + 16);          // K {kb+16..kb+23}
    const _Float16* bp = bRow + k0 + bkb;
    v8h b0 = *(const v8h*)(bp);
    v8h b1 = *(const v8h*)(bp + 16);
    v16h a = __builtin_shufflevector(a0, a1, 0,1,2,3,4,5,6,7,8,9,10,11,12,13,14,15);
    v16h b = __builtin_shufflevector(b0, b1, 0,1,2,3,4,5,6,7,8,9,10,11,12,13,14,15);
    c = __builtin_amdgcn_wmma_f32_16x16x32_f16(
        /*neg_a=*/false, a, /*neg_b=*/false, b,
        /*c_mod=*/(short)0, c, /*reuse_a=*/false, /*reuse_b=*/false);
  }

  int n = n0 + (lane & 15);
  if (n < Nout) {
    float bv = bias ? bias[n] : 0.f;
    int mbase = m0 + half * 8;
#pragma unroll
    for (int i = 0; i < 8; ++i)
      C[(size_t)(mbase + i) * Nout + n] = c[i] + bv;
  }
}

// ---------------------------------------------------------------------------
// fused_pre = T[s] + V[b] + b_fuse ; GELU(exact) ; LayerNorm ; write f16.
// One 256-thread block per (b,s) row.
// ---------------------------------------------------------------------------
__global__ void fuse_ln_kernel(const float* __restrict__ T,
                               const float* __restrict__ V,
                               const float* __restrict__ bfuse,
                               const float* __restrict__ gamma,
                               const float* __restrict__ beta,
                               _Float16* __restrict__ fused) {
  int row = blockIdx.x;                 // b*SS + s
  int s = row & (SS - 1);
  int b = row >> 9;                     // / 512
  int f = threadIdx.x;
  __shared__ float red[256];

  float x = T[(size_t)s * FF + f] + V[(size_t)b * FF + f] + bfuse[f];
  float g = 0.5f * x * (1.0f + erff(x * 0.70710678118654752f));

  red[f] = g; __syncthreads();
  for (int st = 128; st > 0; st >>= 1) { if (f < st) red[f] += red[f + st]; __syncthreads(); }
  float mu = red[0] * (1.0f / FF);
  __syncthreads();
  float d = g - mu;
  red[f] = d * d; __syncthreads();
  for (int st = 128; st > 0; st >>= 1) { if (f < st) red[f] += red[f + st]; __syncthreads(); }
  float var = red[0] * (1.0f / FF);

  float y = d * rsqrtf(var + 1e-5f) * gamma[f] + beta[f];
  fused[(size_t)row * FF + f] = (_Float16)y;
}

// ---------------------------------------------------------------------------
// Zero the loss accumulator slot.
// ---------------------------------------------------------------------------
__global__ void zero_loss_kernel(float* out) {
  if (threadIdx.x == 0 && blockIdx.x == 0) out[BB * SS] = 0.f;
}

// ---------------------------------------------------------------------------
// CRF: forward log-partition + path score (loss) and Viterbi decode.
// One wave32 per batch element; lane = label (lanes 21..31 inactive at -inf).
// Backpointers live in LDS (511*21 bytes). Transition matrix cached in LDS.
// ---------------------------------------------------------------------------
__global__ void crf_kernel(const float* __restrict__ logits,   // [B,S,L]
                           const int*   __restrict__ labels,   // [B,S]
                           const float* __restrict__ start,
                           const float* __restrict__ endw,
                           const float* __restrict__ trans,    // [L,L]
                           float* __restrict__ out) {           // preds then loss
  int b = blockIdx.x;
  int lane = threadIdx.x;
  bool act = lane < LL;

  __shared__ float tr[LL * LL];
  __shared__ unsigned char bp[(SS - 1) * LL];

  for (int i = lane; i < LL * LL; i += 32) tr[i] = trans[i];
  __syncthreads();

  const float* em = logits + (size_t)b * SS * LL;

  float score = act ? (start[lane] + em[lane]) : NEGINF;   // forward scores
  float vit   = score;                                     // viterbi scores

  for (int t = 1; t < SS; ++t) {
    const float* e = em + (size_t)t * LL;
    float fm = NEGINF;           // forward max over prev
    float vm = NEGINF;           // viterbi max over prev
    int   arg = 0;
    // pass 1: maxes (and viterbi argmax)
    for (int i = 0; i < LL; ++i) {
      float sf = __shfl(score, i, 32);
      float sv = __shfl(vit,   i, 32);
      float tij = act ? tr[i * LL + lane] : 0.f;
      float cf = sf + tij;
      if (cf > fm) fm = cf;
      float cv = sv + tij;
      if (cv > vm) { vm = cv; arg = i; }
    }
    // pass 2: sum of exp for forward logsumexp
    float se = 0.f;
    for (int i = 0; i < LL; ++i) {
      float sf = __shfl(score, i, 32);
      float tij = act ? tr[i * LL + lane] : 0.f;
      se += __expf(sf + tij - fm);
    }
    float ev = act ? e[lane] : 0.f;
    score = act ? (fm + logf(se) + ev) : NEGINF;
    vit   = act ? (vm + ev) : NEGINF;
    if (act) bp[(t - 1) * LL + lane] = (unsigned char)arg;
  }
  __syncthreads();

  // logZ = logsumexp over lanes of (score + end)
  float val = act ? (score + endw[lane]) : NEGINF;
  float mx = val;
  for (int off = 16; off > 0; off >>= 1) mx = fmaxf(mx, __shfl_xor(mx, off, 32));
  float se = act ? __expf(val - mx) : 0.f;
  for (int off = 16; off > 0; off >>= 1) se += __shfl_xor(se, off, 32);
  float logZ = mx + logf(se);

  // Viterbi terminal argmax (lowest index on ties, matching jnp.argmax)
  float fv = act ? (vit + endw[lane]) : NEGINF;
  int idx = lane;
  for (int off = 16; off > 0; off >>= 1) {
    float o  = __shfl_xor(fv, off, 32);
    int   oi = __shfl_xor(idx, off, 32);
    if (o > fv || (o == fv && oi < idx)) { fv = o; idx = oi; }
  }

  if (lane == 0) {
    // backtrack
    float* preds = out + (size_t)b * SS;
    int tag = idx;
    preds[SS - 1] = (float)tag;
    for (int t = SS - 2; t >= 0; --t) {
      tag = bp[t * LL + tag];
      preds[t] = (float)tag;
    }
    // numerator (path score of given labels)
    const int* lb = labels + (size_t)b * SS;
    int prev = lb[0];
    float num = start[prev] + em[prev];
    for (int t = 1; t < SS; ++t) {
      int c = lb[t];
      num += em[(size_t)t * LL + c] + tr[prev * LL + c];
      prev = c;
    }
    num += endw[prev];
    float llh = num - logZ;
    atomicAdd(out + BB * SS, -llh * (1.0f / BB));
  }
}

// ---------------------------------------------------------------------------
extern "C" void kernel_launch(void* const* d_in, const int* in_sizes, int n_in,
                              void* d_out, int out_size, void* d_ws, size_t ws_size,
                              hipStream_t stream) {
  const float* visual  = (const float*)d_in[0];   // [64,768]
  const float* text    = (const float*)d_in[1];   // [1,512,768]
  const int*   labels  = (const int*)d_in[2];     // [64,512]
  const float* W_fuse  = (const float*)d_in[3];   // [1536,256]
  const float* b_fuse  = (const float*)d_in[4];   // [256]
  const float* ln_g    = (const float*)d_in[5];   // [256]
  const float* ln_b    = (const float*)d_in[6];   // [256]
  const float* W_out   = (const float*)d_in[7];   // [256,21]
  const float* b_out   = (const float*)d_in[8];   // [21]
  const float* c_start = (const float*)d_in[9];   // [21]
  const float* c_end   = (const float*)d_in[10];  // [21]
  const float* c_trans = (const float*)d_in[11];  // [21,21]
  float* out = (float*)d_out;                      // [64*512 preds] + [loss]

  // Workspace carve-up (all offsets multiples of 256 bytes)
  char* ws = (char*)d_ws;
  size_t o = 0;
  _Float16* t16   = (_Float16*)(ws + o); o += (size_t)SS * DD * 2;       // 768 KB
  _Float16* v16   = (_Float16*)(ws + o); o += (size_t)BB * DD * 2;       //  96 KB
  _Float16* WtT   = (_Float16*)(ws + o); o += (size_t)FF * DD * 2;       // 384 KB
  _Float16* WvT   = (_Float16*)(ws + o); o += (size_t)FF * DD * 2;       // 384 KB
  _Float16* WoT   = (_Float16*)(ws + o); o += (size_t)NPAD * FF * 2;     //  16 KB
  float*    Tm    = (float*)(ws + o);    o += (size_t)SS * FF * 4;       // 512 KB
  float*    Vm    = (float*)(ws + o);    o += (size_t)BB * FF * 4;       //  64 KB
  _Float16* fused = (_Float16*)(ws + o); o += (size_t)BB * SS * FF * 2;  //  16 MB
  float*    logit = (float*)(ws + o);    o += (size_t)BB * SS * LL * 4;  // 2.75 MB

  // 1) L2 normalize (text rows 0..511, visual rows 512..575), emit f16
  l2norm_kernel<<<SS + BB, 256, 0, stream>>>(text, visual, t16, v16);

  // 2) Pre-transpose weights to f16 (contiguous-K operand loads for WMMA)
  prep_wfuse_kernel<<<DD * 2, 256, 0, stream>>>(W_fuse, WtT, WvT);
  prep_wout_kernel<<<FF, 32, 0, stream>>>(W_out, WoT);

  // 3) T = t_norm @ Wt   [512,768]x[768,256]
  wmma_gemm_kernel<<<dim3(SS / 16, FF / 16), 32, 0, stream>>>(
      t16, WtT, Tm, SS, FF, DD, FF, nullptr);
  // 4) V = v_norm @ Wv   [64,768]x[768,256]
  wmma_gemm_kernel<<<dim3(BB / 16, FF / 16), 32, 0, stream>>>(
      v16, WvT, Vm, BB, FF, DD, FF, nullptr);

  // 5) fused = LayerNorm(GELU(T[s] + V[b] + b_fuse))   -> f16 [B*S, 256]
  fuse_ln_kernel<<<BB * SS, 256, 0, stream>>>(Tm, Vm, b_fuse, ln_g, ln_b, fused);

  // 6) logits = fused @ W_out + b_out  [32768,256]x[256,21] (N padded to 32)
  wmma_gemm_kernel<<<dim3((BB * SS) / 16, NPAD / 16), 32, 0, stream>>>(
      fused, WoT, logit, BB * SS, NPAD, FF, LL, b_out);

  // 7) CRF forward (loss) + Viterbi (preds)
  zero_loss_kernel<<<1, 1, 0, stream>>>(out);
  crf_kernel<<<BB, 32, 0, stream>>>(logit, labels, c_start, c_end, c_trans, out);
}
